// CustomCLIP_82411832475982
// MI455X (gfx1250) — matrix operand order repaired and checked by reading, not compiled, not run on hardware
//
#include <hip/hip_runtime.h>
#include <hip/hip_bf16.h>

typedef __attribute__((ext_vector_type(16))) _Float16 v16h;
typedef __attribute__((ext_vector_type(8)))  _Float16 v8h;
typedef __attribute__((ext_vector_type(8)))  float    v8f;

#define NL 12
#define NB 64
#define NC 1000
#define NTPL 8
#define ND 512
#define QROWS (NL*NB)   // 768

// ---------------- WMMA fragment helpers (wave32, 16x16x32 f16) ----------------
// A-matrix 16x32 f16 layout (ISA 7.12.2): lane (g,m) g=lane>>4, m=lane&15 holds
// row m, K elements {k0+g*8+0..7, k0+g*8+16..23}.
__device__ __forceinline__ v16h frag_a(const _Float16* tile, int ld) {
    int lane = threadIdx.x & 31;
    const _Float16* p = tile + (size_t)(lane & 15) * ld + ((lane >> 4) << 3);
    v8h lo = *(const v8h*)p;
    v8h hi = *(const v8h*)(p + 16);
    return __builtin_shufflevector(lo, hi, 0,1,2,3,4,5,6,7,8,9,10,11,12,13,14,15);
}
// B-matrix 32x16 f16: lane (g,n) holds column n, K elements {k0+g*8+0..7, k0+g*8+16..23}.
// Loaded from B^T stored row-major as [N][K] (e.g. weights W[n][k]).
__device__ __forceinline__ v16h frag_b(const _Float16* tileT, int ld) {
    int lane = threadIdx.x & 31;
    const _Float16* p = tileT + (size_t)(lane & 15) * ld + ((lane >> 4) << 3);
    v8h lo = *(const v8h*)p;
    v8h hi = *(const v8h*)(p + 16);
    return __builtin_shufflevector(lo, hi, 0,1,2,3,4,5,6,7,8,9,10,11,12,13,14,15);
}
__device__ __forceinline__ v8f wmma16(v16h a, v16h b, v8f c) {
    return __builtin_amdgcn_wmma_f32_16x16x32_f16(false, a, false, b, (short)0, c, false, false);
}

// ---------------- elementwise f32 -> f16 convert ----------------
__global__ __launch_bounds__(256) void f32_to_f16_kernel(const float* __restrict__ s,
                                                         _Float16* __restrict__ d, int n) {
    int i = blockIdx.x * 256 + threadIdx.x;
    if (i < n) d[i] = (_Float16)s[i];
}

// ---------------- generic WMMA GEMM: C[M,N] = A[M,K] @ Bt[N,K]^T + bias ----------------
__global__ __launch_bounds__(32) void gemm_wmma_kernel(const _Float16* __restrict__ A,
                                                       const _Float16* __restrict__ Bt,
                                                       const float* __restrict__ bias,
                                                       float* __restrict__ Cf32,
                                                       _Float16* __restrict__ Cf16,
                                                       int M, int N, int K) {
    int tn = blockIdx.x, tm = blockIdx.y;
    const _Float16* Ap = A  + (size_t)tm * 16 * K;
    const _Float16* Bp = Bt + (size_t)tn * 16 * K;
    v8f c = {};
    for (int k0 = 0; k0 < K; k0 += 32) {
        v16h a = frag_a(Ap + k0, K);
        v16h b = frag_b(Bp + k0, K);
        c = wmma16(a, b, c);
    }
    int lane = threadIdx.x & 31, n = lane & 15, g = lane >> 4;
    float bv = bias ? bias[tn * 16 + n] : 0.f;
#pragma unroll
    for (int r = 0; r < 8; ++r) {
        int row = tm * 16 + g * 8 + r, col = tn * 16 + n;
        float v = c[r] + bv;
        if (Cf32) Cf32[(size_t)row * N + col] = v;
        if (Cf16) Cf16[(size_t)row * N + col] = (_Float16)v;
    }
}

// ---------------- self-attention over layer axis (tiny: 12x12 per batch) ----------------
__global__ __launch_bounds__(256) void self_attn_kernel(const float* __restrict__ qkv,  // [768,1536] q|k|v
                                                        _Float16* __restrict__ xspre) { // [768,512]
    __shared__ float sc[NL * NL];
    __shared__ float at[NL * NL];
    int b = blockIdx.x, tid = threadIdx.x;
    const float scale = 0.04419417382415922f;  // 1/sqrt(512)
    for (int idx = tid; idx < NL * NL; idx += 256) {
        int l = idx / NL, m = idx % NL;
        const float* qp = qkv + (size_t)(l * NB + b) * 1536;
        const float* kp = qkv + (size_t)(m * NB + b) * 1536 + ND;
        float s = 0.f;
        for (int e = 0; e < ND; ++e) s += qp[e] * kp[e];
        sc[idx] = s * scale;
    }
    __syncthreads();
    if (tid < NL) {
        float mx = -1e30f;
        for (int m = 0; m < NL; ++m) mx = fmaxf(mx, sc[tid * NL + m]);
        float sum = 0.f;
        for (int m = 0; m < NL; ++m) { float e = expf(sc[tid * NL + m] - mx); at[tid * NL + m] = e; sum += e; }
        float inv = 1.f / sum;
        for (int m = 0; m < NL; ++m) at[tid * NL + m] *= inv;
    }
    __syncthreads();
    for (int idx = tid; idx < NL * ND; idx += 256) {
        int l = idx >> 9, d = idx & 511;
        float s = 0.f;
        for (int m = 0; m < NL; ++m) s += at[l * NL + m] * qkv[(size_t)(m * NB + b) * 1536 + 1024 + d];
        xspre[(size_t)(l * NB + b) * ND + d] = (_Float16)s;
    }
}

// ---------------- normalize image features (xs row l=11) ----------------
__global__ __launch_bounds__(256) void img_norm_kernel(const float* __restrict__ xs,
                                                       float* __restrict__ imgn) {
    __shared__ float red[256];
    int b = blockIdx.x, tid = threadIdx.x;
    const float* row = xs + (size_t)(11 * NB + b) * ND;
    float ss = 0.f;
    for (int d = tid; d < ND; d += 256) { float v = row[d]; ss += v * v; }
    red[tid] = ss;
    __syncthreads();
    for (int off = 128; off > 0; off >>= 1) { if (tid < off) red[tid] += red[tid + off]; __syncthreads(); }
    float inv = rsqrtf(red[0]);
    for (int d = tid; d < ND; d += 256) imgn[b * ND + d] = row[d] * inv;
}

// ---------------- fused per-class cross-attention ----------------
// grid = 1000 classes, block = 256 (8 wave32). Loops template-pairs; K/V projected
// on the fly with WMMA; scores 16x16x32 WMMA (2x A-fragment reuse to halve L2
// traffic on q2); softmax + l-reduction scalar; Pbar@V one WMMA per ctx tile;
// out-projection WMMA; writes txt f16 [B,C,D]. Prefetches next template pair.
__global__ __launch_bounds__(256) void cross_attn_kernel(const _Float16* __restrict__ q2,   // [768,512] f16
                                                         const float* __restrict__ text,   // [C,12,8,512] f32
                                                         const _Float16* __restrict__ wci, // cross_in_w f16 [1536,512]
                                                         const float* __restrict__ cib,    // cross_in_b [1536]
                                                         const _Float16* __restrict__ wco, // cross_out_w f16 [512,512]
                                                         const float* __restrict__ cob,    // cross_out_b [512]
                                                         _Float16* __restrict__ txt) {     // [64,1000,512] f16
    extern __shared__ char smem[];
    _Float16* sX    = (_Float16*)smem;             // 32x512 f16 (32KB) — aliases sS
    float*    sS    = (float*)smem;                // 256x32 f32 (32KB)
    _Float16* sKc   = (_Float16*)(smem + 32768);   // 32x512 f16 (32KB) K rows
    _Float16* sVt   = (_Float16*)(smem + 65536);   // 512x32 f16 (32KB) V transposed [d][m]
    float*    sPb   = (float*)(smem + 98304);      // 64x32 f32 (8KB)
    _Float16* sPb16 = (_Float16*)(smem + 106496);  // 64x32 f16 (4KB)
    _Float16* sCtx  = (_Float16*)(smem + 32768);   // 64x512 f16 (64KB) — aliases sKc+sVt

    const int tid  = threadIdx.x;
    const int w    = tid >> 5;
    const int lane = tid & 31;
    const int n    = lane & 15;
    const int g    = lane >> 4;
    const int cls  = blockIdx.x;
    const float scale = 0.04419417382415922f;  // 1/sqrt(512)

    v8f acc[16];  // ctx accumulators: this wave owns d-tiles nt=w*4..w*4+3 x all 4 b-tiles
#pragma unroll
    for (int i = 0; i < 16; ++i) acc[i] = {};

    for (int tp = 0; tp < 4; ++tp) {  // template pairs (t0=2tp, t1=2tp+1)
        const int t0 = tp * 2;
        __syncthreads();
        // (a) stage text rows to LDS f16; rows 0..11 -> t0, 16..27 -> t1, pads zero.
        //     Also prefetch the next template pair (global_prefetch path).
        for (int idx = tid; idx < 32 * ND; idx += 256) {
            int r = idx >> 9, col = idx & 511;
            int m = r & 15, ts = r >> 4;
            float v = 0.f;
            if (m < NL) {
                const float* gp = text + (((size_t)cls * NL + m) * NTPL + (t0 + ts)) * ND + col;
                v = *gp;
                if (tp < 3) __builtin_prefetch(gp + 2 * ND, 0, 1);
            }
            sX[idx] = (_Float16)v;
        }
        __syncthreads();
        // (b) project K and V: 64 (matrix,ntile) tasks, 8 per wave; 2 M-tiles each
        for (int i = 0; i < 8; ++i) {
            int tk = w * 8 + i;
            int mat = tk >> 5;      // 0 = K (rows 512..), 1 = V (rows 1024..)
            int nt = tk & 31;
            v8f c0 = {}, c1 = {};
            const _Float16* Wt = wci + (size_t)(512 + mat * 512 + nt * 16) * ND;
            for (int kk = 0; kk < 16; ++kk) {
                v16h bf = frag_b(Wt + kk * 32, ND);
                v16h a0 = frag_a(sX + kk * 32, ND);
                v16h a1 = frag_a(sX + 16 * ND + kk * 32, ND);
                c0 = wmma16(a0, bf, c0);
                c1 = wmma16(a1, bf, c1);
            }
            float bv = cib[512 + mat * 512 + nt * 16 + n];
            if (mat == 0) {
#pragma unroll
                for (int r = 0; r < 8; ++r) {
                    sKc[(g * 8 + r) * ND + nt * 16 + n]        = (_Float16)(c0[r] + bv);
                    sKc[(16 + g * 8 + r) * ND + nt * 16 + n]   = (_Float16)(c1[r] + bv);
                }
            } else {
#pragma unroll
                for (int r = 0; r < 8; ++r) {
                    sVt[(nt * 16 + n) * 32 + g * 8 + r]        = (_Float16)(c0[r] + bv);
                    sVt[(nt * 16 + n) * 32 + 16 + g * 8 + r]   = (_Float16)(c1[r] + bv);
                }
            }
        }
        __syncthreads();
        for (int idx = tid; idx < NB * 32; idx += 256) sPb[idx] = 0.f;
        __syncthreads();
        // (c) scores + softmax + l-reduced probabilities, in 3 chunks of 256 q-rows
        for (int qc = 0; qc < 3; ++qc) {
            const int q0 = qc * 256;
            {   // Q @ K^T : wave handles 2 M-tiles x both N-tiles (each A-frag feeds 2 WMMAs)
                int mtb = w * 2;
                v8f cs[4] = {{}, {}, {}, {}};   // [mi*2 + ntq]
                for (int kk = 0; kk < 16; ++kk) {
                    v16h bf0 = frag_b(sKc + kk * 32, ND);
                    v16h bf1 = frag_b(sKc + 16 * ND + kk * 32, ND);
#pragma unroll
                    for (int mi = 0; mi < 2; ++mi) {
                        v16h af = frag_a(q2 + (size_t)(q0 + (mtb + mi) * 16) * ND + kk * 32, ND);
                        cs[mi * 2 + 0] = wmma16(af, bf0, cs[mi * 2 + 0]);
                        cs[mi * 2 + 1] = wmma16(af, bf1, cs[mi * 2 + 1]);
                    }
                }
#pragma unroll
                for (int mi = 0; mi < 2; ++mi)
#pragma unroll
                    for (int ntq = 0; ntq < 2; ++ntq)
#pragma unroll
                        for (int r = 0; r < 8; ++r)
                            sS[((mtb + mi) * 16 + g * 8 + r) * 32 + ntq * 16 + n] = cs[mi * 2 + ntq][r];
            }
            __syncthreads();
            {   // softmax over m (12) per template; accumulate Pbar[b] = sum_l P / 96
                int qrow = q0 + tid;
                int b = qrow & 63;
#pragma unroll
                for (int half = 0; half < 2; ++half) {
                    int base = half * 16;
                    float e[NL];
                    float mx = -1e30f;
#pragma unroll
                    for (int j = 0; j < NL; ++j) { float s = sS[tid * 32 + base + j] * scale; e[j] = s; mx = fmaxf(mx, s); }
                    float sum = 0.f;
#pragma unroll
                    for (int j = 0; j < NL; ++j) { e[j] = expf(e[j] - mx); sum += e[j]; }
                    float inv = 1.f / (96.f * sum);   // fold 1/(T*L) average
#pragma unroll
                    for (int j = 0; j < NL; ++j) atomicAdd(&sPb[b * 32 + base + j], e[j] * inv);
                }
            }
            __syncthreads();
        }
        for (int idx = tid; idx < NB * 32; idx += 256) sPb16[idx] = (_Float16)sPb[idx];
        __syncthreads();
        // (d) ctx += Pbar[64x32] @ V[32x512] : one WMMA per tile
#pragma unroll
        for (int jt = 0; jt < 4; ++jt) {
            int nt = w * 4 + jt;
            v16h bf = frag_b(sVt + (size_t)nt * 16 * 32, 32);
#pragma unroll
            for (int mt = 0; mt < 4; ++mt) {
                v16h af = frag_a(sPb16 + mt * 16 * 32, 32);
                acc[jt * 4 + mt] = wmma16(af, bf, acc[jt * 4 + mt]);
            }
        }
        __syncthreads();
    }
    // stage ctx into LDS f16 (reuses sKc|sVt)
#pragma unroll
    for (int jt = 0; jt < 4; ++jt) {
        int nt = w * 4 + jt;
#pragma unroll
        for (int mt = 0; mt < 4; ++mt)
#pragma unroll
            for (int r = 0; r < 8; ++r)
                sCtx[(mt * 16 + g * 8 + r) * ND + nt * 16 + n] = (_Float16)acc[jt * 4 + mt][r];
    }
    __syncthreads();
    // out-projection: txt[b,cls,:] = ctx @ cross_out_w^T + b
    for (int jt = 0; jt < 4; ++jt) {
        int nt = w * 4 + jt;
        v8f co[4] = {{}, {}, {}, {}};
        const _Float16* Wo = wco + (size_t)nt * 16 * ND;
        for (int kk = 0; kk < 16; ++kk) {
            v16h bf = frag_b(Wo + kk * 32, ND);
#pragma unroll
            for (int mt = 0; mt < 4; ++mt) {
                v16h af = frag_a(sCtx + mt * 16 * ND + kk * 32, ND);
                co[mt] = wmma16(af, bf, co[mt]);
            }
        }
        float bv = cob[nt * 16 + n];
#pragma unroll
        for (int mt = 0; mt < 4; ++mt)
#pragma unroll
            for (int r = 0; r < 8; ++r) {
                int brow = mt * 16 + g * 8 + r;
                txt[((size_t)brow * NC + cls) * ND + nt * 16 + n] = (_Float16)(co[mt][r] + bv);
            }
    }
}

// ---------------- per-(b,d) norm over classes, fused with img_n weighting ----------------
__global__ __launch_bounds__(256) void denom_kernel(const _Float16* __restrict__ txt,
                                                    const float* __restrict__ imgn,
                                                    float* __restrict__ wvec) {
    int b = blockIdx.x, tid = threadIdx.x;
    for (int dd = 0; dd < 2; ++dd) {
        int d = tid + dd * 256;
        const _Float16* p = txt + (size_t)b * NC * ND + d;
        float ss = 0.f;
        for (int c = 0; c < NC; ++c) { float v = (float)p[(size_t)c * ND]; ss += v * v; }
        wvec[b * ND + d] = imgn[b * ND + d] * rsqrtf(ss);
    }
}

// ---------------- logits[b,c] = exp(ls) * sum_d wvec[b,d]*txt[b,c,d] ----------------
__global__ __launch_bounds__(256) void logits_kernel(const _Float16* __restrict__ txt,
                                                     const float* __restrict__ wvec,
                                                     const float* __restrict__ logit_scale,
                                                     float* __restrict__ out) {
    int b = blockIdx.x;
    int w = threadIdx.x >> 5, lane = threadIdx.x & 31;
    int c = blockIdx.y * 8 + w;
    const _Float16* p = txt + ((size_t)b * NC + c) * ND;
    const float* wv = wvec + b * ND;
    float s = 0.f;
    for (int j = 0; j < 16; ++j) { int d = lane + j * 32; s += wv[d] * (float)p[d]; }
    for (int off = 16; off > 0; off >>= 1) s += __shfl_down(s, off);
    if (lane == 0) out[(size_t)b * NC + c] = expf(logit_scale[0]) * s;
}

extern "C" void kernel_launch(void* const* d_in, const int* in_sizes, int n_in,
                              void* d_out, int out_size, void* d_ws, size_t ws_size,
                              hipStream_t stream) {
    (void)in_sizes; (void)n_in; (void)out_size; (void)ws_size;
    const float* x    = (const float*)d_in[0];
    const float* text = (const float*)d_in[1];
    const float* siw  = (const float*)d_in[2];
    const float* sib  = (const float*)d_in[3];
    const float* sow  = (const float*)d_in[4];
    const float* sob  = (const float*)d_in[5];
    const float* ciw  = (const float*)d_in[6];
    const float* cib  = (const float*)d_in[7];
    const float* cow  = (const float*)d_in[8];
    const float* cob  = (const float*)d_in[9];
    const float* ls   = (const float*)d_in[10];
    float* out = (float*)d_out;

    char* ws = (char*)d_ws;
    size_t off = 0;
    auto alloc = [&](size_t bytes) -> char* {
        char* p = ws + off;
        off += (bytes + 255) & ~(size_t)255;
        return p;
    };
    _Float16* x16   = (_Float16*)alloc((size_t)QROWS * ND * 2);
    _Float16* siw16 = (_Float16*)alloc((size_t)1536 * ND * 2);
    _Float16* sow16 = (_Float16*)alloc((size_t)ND * ND * 2);
    _Float16* ciw16 = (_Float16*)alloc((size_t)1536 * ND * 2);
    _Float16* cow16 = (_Float16*)alloc((size_t)ND * ND * 2);
    float*    qkv   = (float*)alloc((size_t)QROWS * 1536 * 4);
    _Float16* xspre = (_Float16*)alloc((size_t)QROWS * ND * 2);
    float*    xsf   = (float*)alloc((size_t)QROWS * ND * 4);
    _Float16* xs16  = (_Float16*)alloc((size_t)QROWS * ND * 2);
    _Float16* q216  = (_Float16*)alloc((size_t)QROWS * ND * 2);
    float*    imgn  = (float*)alloc((size_t)NB * ND * 4);
    float*    wvec  = (float*)alloc((size_t)NB * ND * 4);
    _Float16* txt   = (_Float16*)alloc((size_t)NB * NC * ND * 2);

    auto cvt = [&](const float* s, _Float16* d, int n) {
        f32_to_f16_kernel<<<(n + 255) / 256, 256, 0, stream>>>(s, d, n);
    };
    cvt(x, x16, QROWS * ND);
    cvt(siw, siw16, 1536 * ND);
    cvt(sow, sow16, ND * ND);
    cvt(ciw, ciw16, 1536 * ND);
    cvt(cow, cow16, ND * ND);

    // qkv = x @ self_in_w^T + b    [768,1536]
    gemm_wmma_kernel<<<dim3(1536 / 16, QROWS / 16), 32, 0, stream>>>(x16, siw16, sib, qkv, nullptr, QROWS, 1536, ND);
    // tiny self-attention (softmax over L=12) -> xspre f16 [768,512]
    self_attn_kernel<<<NB, 256, 0, stream>>>(qkv, xspre);
    // xs = xspre @ self_out_w^T + b  (f32 for norm, f16 for next GEMM)
    gemm_wmma_kernel<<<dim3(ND / 16, QROWS / 16), 32, 0, stream>>>(xspre, sow16, sob, xsf, xs16, QROWS, ND, ND);
    // q2 = xs @ cross_Wq^T + bq  (f16)
    gemm_wmma_kernel<<<dim3(ND / 16, QROWS / 16), 32, 0, stream>>>(xs16, ciw16, cib, nullptr, q216, QROWS, ND, ND);
    // normalized image features
    img_norm_kernel<<<NB, 256, 0, stream>>>(xsf, imgn);
    // big fused per-class cross-attention (dominant WMMA work)
    cross_attn_kernel<<<NC, 256, 110592, stream>>>(q216, text, ciw16, cib, cow16, cob, txt);
    // norm over classes per (b,d), fused with img weighting
    denom_kernel<<<NB, 256, 0, stream>>>(txt, imgn, wvec);
    // final logits [64,1000]
    logits_kernel<<<dim3(NB, NC / 8), 256, 0, stream>>>(txt, wvec, ls, out);
}
